// BeliefStateTransformer_82643760710014
// MI455X (gfx1250) — compile-verified, matrix operand order checked by reading, not run on hardware
//
#include <hip/hip_runtime.h>
#include <hip/hip_bf16.h>
#include <math.h>

// ---------------------------------------------------------------------------
// BeliefStateTransformer forward for MI455X (gfx1250, wave32, WMMA + TDM).
// B=2, L=1024, D=1024, H=16, hd=64, NL=8, FF=4096.
// GEMMs + attention on v_wmma_f32_16x16x32_f16 (f32 converted to f16 on the
// fly, f32 accumulation). GEMM is double-buffered: A tiles DMA'd by the
// Tensor Data Mover (tensor_load_to_lds / TENSORcnt), B tiles staged through
// VALU convert+transpose with global_prefetch warming the next tile.
// ---------------------------------------------------------------------------

#define BQ   2
#define LQ   1024
#define DQ   1024
#define HQ   16
#define HDQ  64
#define NLQ  8
#define FFQ  4096
#define MROWS (BQ * LQ)          // 2048

typedef float        v8f   __attribute__((ext_vector_type(8)));
typedef _Float16     v16h  __attribute__((ext_vector_type(16)));
typedef _Float16     v8h   __attribute__((ext_vector_type(8)));
typedef _Float16     v4h   __attribute__((ext_vector_type(4)));
typedef unsigned int u32x4 __attribute__((ext_vector_type(4)));
typedef unsigned int u32x8 __attribute__((ext_vector_type(8)));

static __device__ __forceinline__ v8f zero8() {
  v8f z = {0.f,0.f,0.f,0.f,0.f,0.f,0.f,0.f};
  return z;
}

static __device__ __forceinline__ void wait_tensor0() {
#if __has_builtin(__builtin_amdgcn_s_wait_tensorcnt)
  __builtin_amdgcn_s_wait_tensorcnt(0);
#else
  asm volatile("s_wait_tensorcnt 0x0" ::: "memory");
#endif
}

// ---------------------------------------------------------------------------
// Fragment loaders.
// CDNA5 16-bit A-matrix 16x32 layout (ISA 7.12.2): lane l holds row m=l%16;
// lanes 0-15: VGPR0-3 = K0..7, VGPR4-7 = K16..23; lanes 16-31: K8..15,K24..31.
// B fragments use the mirrored layout (lane = column), so the same loader
// works for B tiles staged transposed (Bt[n][k]).
// ---------------------------------------------------------------------------
static __device__ __forceinline__ v16h load_frag_lds(const _Float16* base,
                                                     int row0, int ld) {
  const int l  = threadIdx.x & 31;
  const int r  = row0 + (l & 15);
  const int k0 = (l & 16) ? 8 : 0;
  const _Float16* p = base + r * ld + k0;
  union { v16h v; struct { v8h lo, hi; } s; } u;
  u.s.lo = *(const v8h*)(p);        // ds_load_b128
  u.s.hi = *(const v8h*)(p + 16);   // ds_load_b128
  return u.v;
}

// Same fragment pattern from row-major f32 (global or LDS), convert + scale.
static __device__ __forceinline__ v16h load_frag_f32(const float* base,
                                                     int row0, int ld,
                                                     float scale) {
  const int l  = threadIdx.x & 31;
  const int r  = row0 + (l & 15);
  const int k0 = (l & 16) ? 8 : 0;
  const float* p = base + (long)r * ld + k0;
  float4 a0 = *(const float4*)(p);
  float4 a1 = *(const float4*)(p + 4);
  float4 b0 = *(const float4*)(p + 16);
  float4 b1 = *(const float4*)(p + 20);
  v16h f;
  f[0]=(_Float16)(a0.x*scale); f[1]=(_Float16)(a0.y*scale);
  f[2]=(_Float16)(a0.z*scale); f[3]=(_Float16)(a0.w*scale);
  f[4]=(_Float16)(a1.x*scale); f[5]=(_Float16)(a1.y*scale);
  f[6]=(_Float16)(a1.z*scale); f[7]=(_Float16)(a1.w*scale);
  f[8]=(_Float16)(b0.x*scale); f[9]=(_Float16)(b0.y*scale);
  f[10]=(_Float16)(b0.z*scale); f[11]=(_Float16)(b0.w*scale);
  f[12]=(_Float16)(b1.x*scale); f[13]=(_Float16)(b1.y*scale);
  f[14]=(_Float16)(b1.z*scale); f[15]=(_Float16)(b1.w*scale);
  return f;
}

// ---------------------------------------------------------------------------
// Tensor Data Mover: DMA one 128-row x 32-col f32 tile of A into LDS.
// D# per CDNA5 ISA: group0 = {count/type/addresses},
// group1 = {data_size, tensor dims, tile dims, strides}. 2-D tile so
// VADDR2/VADDR3 are omitted (2-operand asm form). Issued by one wave;
// completion via TENSORcnt.
// ---------------------------------------------------------------------------
static __device__ __forceinline__ void tdm_load_A_tile(const float* A, int M,
                                                       int K, int bm, int k0,
                                                       float* dstAf) {
  const unsigned long long gaddr =
      (unsigned long long)(const void*)(A + (long)bm * K + k0);
  const unsigned int laddr = (unsigned int)(unsigned long long)dstAf;
  u32x4 g0;
  g0[0] = 1u;                                            // count=1 (valid D#)
  g0[1] = laddr;                                         // lds_addr
  g0[2] = (unsigned int)(gaddr & 0xffffffffu);           // global_addr[31:0]
  g0[3] = (unsigned int)((gaddr >> 32) & 0x01ffffffu)    // global_addr[56:32]
          | (2u << 30);                                  // type=2 ("image")
  const unsigned int td0 = (unsigned int)K;              // tensor_dim0
  const unsigned int td1 = (unsigned int)M;              // tensor_dim1
  const unsigned long long st0 = (unsigned long long)K;  // dim0 stride
  u32x8 g1;
  g1[0] = (2u << 16);                        // workgroup_mask=0, data_size=4B
  g1[1] = (td0 & 0xffffu) << 16;             // tensor_dim0[15:0] @ bits 63:48
  g1[2] = (td0 >> 16) | ((td1 & 0xffffu) << 16);
  g1[3] = (td1 >> 16) | (32u << 16);         // tile_dim0 = 32 @ bits 127:112
  g1[4] = 128u;                              // tile_dim1 = 128
  g1[5] = (unsigned int)(st0 & 0xffffffffu); // tensor_dim0_stride[31:0]
  g1[6] = (unsigned int)(st0 >> 32);         // stride[47:32]; dim1_stride=0
  g1[7] = 0u;
  asm volatile("tensor_load_to_lds %0, %1" :: "s"(g0), "s"(g1) : "memory");
}

// ---------------------------------------------------------------------------
// Stage a 32(k) x 128(n) f32 weight tile, transposed+converted -> Bt[n][k] f16.
// Also warms the following k-tile with global_prefetch.
// ---------------------------------------------------------------------------
static __device__ __forceinline__ void stage_B_tile(const float* W, int N,
                                                    int K, int bn, int k0,
                                                    _Float16* dstBt, int tid) {
#pragma unroll
  for (int j = 0; j < 4; ++j) {
    const int p  = tid + 256 * j;
    const int kr = p >> 5;               // 32 float4 per 128-col row
    const int nc = (p & 31) * 4;
    float4 v = *(const float4*)(W + (long)(k0 + kr) * N + bn + nc);
    dstBt[(nc + 0) * 32 + kr] = (_Float16)v.x;
    dstBt[(nc + 1) * 32 + kr] = (_Float16)v.y;
    dstBt[(nc + 2) * 32 + kr] = (_Float16)v.z;
    dstBt[(nc + 3) * 32 + kr] = (_Float16)v.w;
    if (k0 + 32 < K)                      // warm the next k-tile (speculative)
      __builtin_prefetch((const void*)(W + (long)(k0 + 32 + kr) * N + bn + nc),
                         0, 1);
  }
}

// ---------------------------------------------------------------------------
// Embedding + input projection + positional encoding.
// ---------------------------------------------------------------------------
__global__ __launch_bounds__(256)
void embed_kernel(const int* __restrict__ card_ids,
                  const int* __restrict__ action_ids,
                  const float* __restrict__ bet_amounts,
                  const float* __restrict__ card_tab,
                  const float* __restrict__ action_tab,
                  const float* __restrict__ bet_W,
                  const float* __restrict__ bet_b,
                  const float* __restrict__ in_W,
                  const float* __restrict__ in_b,
                  float* __restrict__ x) {
  const int idx = blockIdx.x * 256 + threadIdx.x;   // over MROWS*D
  const int d   = idx & (DQ - 1);
  const int t   = idx >> 10;
  const int pos = t & (LQ - 1);
  const int ci  = card_ids[t];
  const int ai  = action_ids[t];
  const float ba = bet_amounts[t];
  float acc = in_b[d];
#pragma unroll 4
  for (int j = 0; j < 16; ++j)
    acc += card_tab[ci * 16 + j] * in_W[j * DQ + d];
#pragma unroll 4
  for (int j = 0; j < 16; ++j)
    acc += action_tab[ai * 16 + j] * in_W[(16 + j) * DQ + d];
#pragma unroll 4
  for (int j = 0; j < 16; ++j)
    acc += (ba * bet_W[j] + bet_b[j]) * in_W[(32 + j) * DQ + d];
  const int   i2  = d & ~1;
  const float div = __expf((float)i2 * (-9.210340371976184f / (float)DQ));
  const float ang = (float)pos * div;
  acc += (d & 1) ? __cosf(ang) : __sinf(ang);
  x[idx] = acc;
}

// ---------------------------------------------------------------------------
// WMMA GEMM: C[M,N] = act(A[M,K] @ W[K,N] + bias[N])
// Block tile 128x128x32, 8 waves (2x4), each wave 64x32 = 8 WMMA accums.
// Double-buffered: A via TDM (f32 LDS), B via VALU convert/transpose (f16).
// ---------------------------------------------------------------------------
template <bool RELU>
__global__ __launch_bounds__(256)
void wmma_gemm(const float* __restrict__ A, const float* __restrict__ W,
               const float* __restrict__ bias, float* __restrict__ C,
               int M, int N, int K) {
  __shared__ float    Af[2][128 * 32];   // TDM destination (f32)
  __shared__ _Float16 Bt[2][128 * 32];   // Bt[n][k] (f16)
  const int tid  = threadIdx.x;
  const int wave = tid >> 5;
  const int wr   = wave >> 2;            // 0..1  (M direction)
  const int wc   = wave & 3;             // 0..3  (N direction)
  const int bm   = blockIdx.y * 128;
  const int bn   = blockIdx.x * 128;
  const int nk   = K >> 5;

  v8f acc[4][2];
#pragma unroll
  for (int mt = 0; mt < 4; ++mt) { acc[mt][0] = zero8(); acc[mt][1] = zero8(); }

  // Prologue: fill buffer 0.
  stage_B_tile(W, N, K, bn, 0, &Bt[0][0], tid);
  if (tid < 32) {                        // wave 0 drives the TDM
    tdm_load_A_tile(A, M, K, bm, 0, &Af[0][0]);
    wait_tensor0();
  }
  __syncthreads();

  for (int ks = 0; ks < nk; ++ks) {
    const int cur = ks & 1;
    const int nxt = cur ^ 1;
    if (ks + 1 < nk) {                   // prefetch next tile into other buffer
      stage_B_tile(W, N, K, bn, (ks + 1) * 32, &Bt[nxt][0], tid);
      if (tid < 32)
        tdm_load_A_tile(A, M, K, bm, (ks + 1) * 32, &Af[nxt][0]);
    }

    v16h bf0 = load_frag_lds(&Bt[cur][0], wc * 32, 32);
    v16h bf1 = load_frag_lds(&Bt[cur][0], wc * 32 + 16, 32);
#pragma unroll
    for (int mt = 0; mt < 4; ++mt) {
      v16h af = load_frag_f32(&Af[cur][0], wr * 64 + mt * 16, 32, 1.f);
      acc[mt][0] = __builtin_amdgcn_wmma_f32_16x16x32_f16(
          false, af, false, bf0, (short)0, acc[mt][0], false, false);
      acc[mt][1] = __builtin_amdgcn_wmma_f32_16x16x32_f16(
          false, af, false, bf1, (short)0, acc[mt][1], false, false);
    }

    if (ks + 1 < nk) {
      if (tid < 32) wait_tensor0();      // TDM of next tile complete
      __syncthreads();                   // publish both staged buffers
    }
  }

  // Epilogue: C/D layout -> lane l holds (m = i + 8*(l>>4), n = l&15).
  const int l    = tid & 31;
  const int crow = (l & 16) ? 8 : 0;
  const int ccol = l & 15;
#pragma unroll
  for (int mt = 0; mt < 4; ++mt) {
#pragma unroll
    for (int nt = 0; nt < 2; ++nt) {
      const int col = bn + wc * 32 + nt * 16 + ccol;
      const float bb = bias[col];
#pragma unroll
      for (int i = 0; i < 8; ++i) {
        const int row = bm + wr * 64 + mt * 16 + i + crow;
        float r = acc[mt][nt][i] + bb;
        if (RELU) r = fmaxf(r, 0.f);
        C[(long)row * N + col] = r;
      }
    }
  }
}

// ---------------------------------------------------------------------------
// Flash attention (causal). qkv layout: [B*L][3*D], cols = [Q | K | V],
// head h occupies cols h*64..h*64+63 within each third.
// Block = 128 threads (4 waves) handles one 64-query tile of one (b,h).
// ---------------------------------------------------------------------------
__global__ __launch_bounds__(128)
void flash_attn(const float* __restrict__ qkv, float* __restrict__ out) {
  const int qb   = blockIdx.x;          // query block 0..15
  const int bh   = blockIdx.y;          // 0..31
  const int b    = bh / HQ;
  const int h    = bh % HQ;
  const int wave = threadIdx.x >> 5;
  const int l    = threadIdx.x & 31;
  const int crow = (l & 16) ? 8 : 0;
  const int ccol = l & 15;
  const int ldq  = 3 * DQ;

  const float* base = qkv + (long)b * LQ * ldq;
  const float* Qb   = base + 0 * DQ + h * HDQ;
  const float* Kb   = base + 1 * DQ + h * HDQ;
  const float* Vb   = base + 2 * DQ + h * HDQ;

  __shared__ _Float16 Vt[HDQ * 80];        // Vt[hd][key], padded ld=80
  __shared__ _Float16 Pl[4][16 * 64];      // per-wave P tile [m][key]

  const float qscale = 0.125f;             // 1/sqrt(64)
  v16h qf0 = load_frag_f32(Qb + 0,  qb * 64 + wave * 16, ldq, qscale);
  v16h qf1 = load_frag_f32(Qb + 32, qb * 64 + wave * 16, ldq, qscale);

  v8f   o[4];
  float rmax[8], rsum[8];
#pragma unroll
  for (int nt = 0; nt < 4; ++nt) o[nt] = zero8();
#pragma unroll
  for (int i = 0; i < 8; ++i) { rmax[i] = -3.0e38f; rsum[i] = 0.f; }

  for (int kb = 0; kb <= qb; ++kb) {
    __syncthreads();   // protect Vt from previous iteration's readers
    // Stage V block transposed: V[key][hd] -> Vt[hd][key].
#pragma unroll
    for (int j = 0; j < 8; ++j) {
      const int p   = threadIdx.x + 128 * j;
      const int key = p >> 4;
      const int hc  = (p & 15) * 4;
      float4 v = *(const float4*)(Vb + (long)(kb * 64 + key) * ldq + hc);
      Vt[(hc + 0) * 80 + key] = (_Float16)v.x;
      Vt[(hc + 1) * 80 + key] = (_Float16)v.y;
      Vt[(hc + 2) * 80 + key] = (_Float16)v.z;
      Vt[(hc + 3) * 80 + key] = (_Float16)v.w;
      if (kb < qb)   // warm next key-block of V
        __builtin_prefetch(
            (const void*)(Vb + (long)((kb + 1) * 64 + key) * ldq + hc), 0, 1);
    }
    __syncthreads();

    // S = Q @ K^T (K rows from global are exactly B-fragment chunks).
    v8f s[4];
#pragma unroll
    for (int nt = 0; nt < 4; ++nt) {
      v16h kf0 = load_frag_f32(Kb + 0,  kb * 64 + nt * 16, ldq, 1.f);
      v16h kf1 = load_frag_f32(Kb + 32, kb * 64 + nt * 16, ldq, 1.f);
      s[nt] = zero8();
      s[nt] = __builtin_amdgcn_wmma_f32_16x16x32_f16(
          false, qf0, false, kf0, (short)0, s[nt], false, false);
      s[nt] = __builtin_amdgcn_wmma_f32_16x16x32_f16(
          false, qf1, false, kf1, (short)0, s[nt], false, false);
    }

    // Causal mask on the diagonal block.
    if (kb == qb) {
#pragma unroll
      for (int nt = 0; nt < 4; ++nt)
#pragma unroll
        for (int i = 0; i < 8; ++i) {
          const int qrow = wave * 16 + i + crow;
          const int kcol = nt * 16 + ccol;
          if (kcol > qrow) s[nt][i] = -3.0e38f;
        }
    }

    // Online softmax: row reductions = butterflies within 16-lane groups.
#pragma unroll
    for (int i = 0; i < 8; ++i) {
      float mnew = s[0][i];
      mnew = fmaxf(mnew, s[1][i]);
      mnew = fmaxf(mnew, s[2][i]);
      mnew = fmaxf(mnew, s[3][i]);
      mnew = fmaxf(mnew, __shfl_xor(mnew, 1, 32));
      mnew = fmaxf(mnew, __shfl_xor(mnew, 2, 32));
      mnew = fmaxf(mnew, __shfl_xor(mnew, 4, 32));
      mnew = fmaxf(mnew, __shfl_xor(mnew, 8, 32));
      const float mtot = fmaxf(rmax[i], mnew);
      const float corr = __expf(rmax[i] - mtot);
      rmax[i] = mtot;
      float ssum = 0.f;
#pragma unroll
      for (int nt = 0; nt < 4; ++nt) {
        const float pv = __expf(s[nt][i] - mtot);
        s[nt][i] = pv;
        ssum += pv;
      }
      ssum += __shfl_xor(ssum, 1, 32);
      ssum += __shfl_xor(ssum, 2, 32);
      ssum += __shfl_xor(ssum, 4, 32);
      ssum += __shfl_xor(ssum, 8, 32);
      rsum[i] = rsum[i] * corr + ssum;
#pragma unroll
      for (int nt = 0; nt < 4; ++nt) o[nt][i] *= corr;
    }

    // P (C-layout f32) -> LDS f16 -> A-layout fragments.
#pragma unroll
    for (int nt = 0; nt < 4; ++nt)
#pragma unroll
      for (int i = 0; i < 8; ++i)
        Pl[wave][(i + crow) * 64 + nt * 16 + ccol] = (_Float16)s[nt][i];
    __syncthreads();

    // O += P @ V  (key contraction 64 = 2 chunks of 32).
#pragma unroll
    for (int c = 0; c < 2; ++c) {
      v16h pf = load_frag_lds(&Pl[wave][c * 32], 0, 64);
#pragma unroll
      for (int nt = 0; nt < 4; ++nt) {
        v16h vf = load_frag_lds(Vt + c * 32, nt * 16, 80);
        o[nt] = __builtin_amdgcn_wmma_f32_16x16x32_f16(
            false, pf, false, vf, (short)0, o[nt], false, false);
      }
    }
  }

  // Normalize and write concat(head) output: out[t][h*64 + d].
#pragma unroll
  for (int nt = 0; nt < 4; ++nt)
#pragma unroll
    for (int i = 0; i < 8; ++i) {
      const int q = qb * 64 + wave * 16 + i + crow;
      const int t = b * LQ + q;
      const int col = h * HDQ + nt * 16 + ccol;
      out[(long)t * DQ + col] = o[nt][i] / rsum[i];
    }
}

// ---------------------------------------------------------------------------
// Fused residual add + LayerNorm: x[row] = LN(x[row] + t[row]) * g + b.
// ---------------------------------------------------------------------------
__global__ __launch_bounds__(256)
void add_ln(float* __restrict__ x, const float* __restrict__ t,
            const float* __restrict__ g, const float* __restrict__ beta) {
  const int row = blockIdx.x;
  __shared__ float red1[8];
  __shared__ float red2[8];
  float v[4];
  float s = 0.f;
#pragma unroll
  for (int j = 0; j < 4; ++j) {
    const int c = threadIdx.x + 256 * j;
    v[j] = x[(long)row * DQ + c] + t[(long)row * DQ + c];
    s += v[j];
  }
  s += __shfl_xor(s, 1, 32);  s += __shfl_xor(s, 2, 32);
  s += __shfl_xor(s, 4, 32);  s += __shfl_xor(s, 8, 32);
  s += __shfl_xor(s, 16, 32);
  if ((threadIdx.x & 31) == 0) red1[threadIdx.x >> 5] = s;
  __syncthreads();
  float tot = 0.f;
#pragma unroll
  for (int w = 0; w < 8; ++w) tot += red1[w];
  const float mu = tot * (1.0f / DQ);

  float vs = 0.f;
#pragma unroll
  for (int j = 0; j < 4; ++j) { const float dd = v[j] - mu; vs += dd * dd; }
  vs += __shfl_xor(vs, 1, 32);  vs += __shfl_xor(vs, 2, 32);
  vs += __shfl_xor(vs, 4, 32);  vs += __shfl_xor(vs, 8, 32);
  vs += __shfl_xor(vs, 16, 32);
  if ((threadIdx.x & 31) == 0) red2[threadIdx.x >> 5] = vs;
  __syncthreads();
  float vtot = 0.f;
#pragma unroll
  for (int w = 0; w < 8; ++w) vtot += red2[w];
  const float rstd = rsqrtf(vtot * (1.0f / DQ) + 1e-5f);

#pragma unroll
  for (int j = 0; j < 4; ++j) {
    const int c = threadIdx.x + 256 * j;
    x[(long)row * DQ + c] = (v[j] - mu) * rstd * g[c] + beta[c];
  }
}

// ---------------------------------------------------------------------------
// Extract belief = x[:, L-1, :]  -> out[B*D].
// ---------------------------------------------------------------------------
__global__ __launch_bounds__(256)
void extract_last(const float* __restrict__ x, float* __restrict__ out) {
  const int idx = blockIdx.x * 256 + threadIdx.x;  // 0..B*D-1
  const int b = idx >> 10;
  const int d = idx & (DQ - 1);
  out[idx] = x[(long)(b * LQ + (LQ - 1)) * DQ + d];
}

// ---------------------------------------------------------------------------
// Host launcher.
// ---------------------------------------------------------------------------
extern "C" void kernel_launch(void* const* d_in, const int* in_sizes, int n_in,
                              void* d_out, int out_size, void* d_ws, size_t ws_size,
                              hipStream_t stream) {
  (void)in_sizes; (void)n_in; (void)out_size; (void)ws_size;
  const int*   card_ids    = (const int*)  d_in[0];
  const int*   action_ids  = (const int*)  d_in[1];
  const float* bet_amounts = (const float*)d_in[2];
  const float* card_tab    = (const float*)d_in[3];
  const float* action_tab  = (const float*)d_in[4];
  const float* bet_W       = (const float*)d_in[5];
  const float* bet_b       = (const float*)d_in[6];
  const float* in_W        = (const float*)d_in[7];
  const float* in_b        = (const float*)d_in[8];
  const float* qkv_W       = (const float*)d_in[9];
  const float* qkv_b       = (const float*)d_in[10];
  const float* out_W       = (const float*)d_in[11];
  const float* out_b       = (const float*)d_in[12];
  const float* ln1_g       = (const float*)d_in[13];
  const float* ln1_b       = (const float*)d_in[14];
  const float* ffn_W1      = (const float*)d_in[15];
  const float* ffn_b1      = (const float*)d_in[16];
  const float* ffn_W2      = (const float*)d_in[17];
  const float* ffn_b2      = (const float*)d_in[18];
  const float* ln2_g       = (const float*)d_in[19];
  const float* ln2_b       = (const float*)d_in[20];

  // Workspace layout (floats): x | qkv | abuf | hbuf  = 72 MB total.
  float* ws   = (float*)d_ws;
  float* x    = ws;                                   // 2048*1024
  float* qkv  = x    + (size_t)MROWS * DQ;            // 2048*3072
  float* abuf = qkv  + (size_t)MROWS * 3 * DQ;        // 2048*1024
  float* hbuf = abuf + (size_t)MROWS * DQ;            // 2048*4096

  embed_kernel<<<(MROWS * DQ) / 256, 256, 0, stream>>>(
      card_ids, action_ids, bet_amounts, card_tab, action_tab,
      bet_W, bet_b, in_W, in_b, x);

  for (int i = 0; i < NLQ; ++i) {
    const float* qW = qkv_W  + (size_t)i * DQ * 3 * DQ;
    const float* qB = qkv_b  + (size_t)i * 3 * DQ;
    const float* oW = out_W  + (size_t)i * DQ * DQ;
    const float* oB = out_b  + (size_t)i * DQ;
    const float* w1 = ffn_W1 + (size_t)i * DQ * FFQ;
    const float* b1 = ffn_b1 + (size_t)i * FFQ;
    const float* w2 = ffn_W2 + (size_t)i * FFQ * DQ;
    const float* b2 = ffn_b2 + (size_t)i * DQ;

    // qkv = x @ qkv_W + qkv_b            (2048 x 3072 x 1024)
    wmma_gemm<false><<<dim3(3 * DQ / 128, MROWS / 128), 256, 0, stream>>>(
        x, qW, qB, qkv, MROWS, 3 * DQ, DQ);
    // abuf = attention(qkv)
    flash_attn<<<dim3(LQ / 64, BQ * HQ), 128, 0, stream>>>(qkv, abuf);
    // hbuf[:, :D] = abuf @ out_W + out_b (2048 x 1024 x 1024)
    wmma_gemm<false><<<dim3(DQ / 128, MROWS / 128), 256, 0, stream>>>(
        abuf, oW, oB, hbuf, MROWS, DQ, DQ);
    // x = LN(x + hbuf)
    add_ln<<<MROWS, 256, 0, stream>>>(x, hbuf, ln1_g + (size_t)i * DQ,
                                      ln1_b + (size_t)i * DQ);
    // hbuf = relu(x @ W1 + b1)           (2048 x 4096 x 1024)
    wmma_gemm<true><<<dim3(FFQ / 128, MROWS / 128), 256, 0, stream>>>(
        x, w1, b1, hbuf, MROWS, FFQ, DQ);
    // abuf = hbuf @ W2 + b2              (2048 x 1024 x 4096)
    wmma_gemm<false><<<dim3(DQ / 128, MROWS / 128), 256, 0, stream>>>(
        hbuf, w2, b2, abuf, MROWS, DQ, FFQ);
    // x = LN(x + abuf)
    add_ln<<<MROWS, 256, 0, stream>>>(x, abuf, ln2_g + (size_t)i * DQ,
                                      ln2_b + (size_t)i * DQ);
  }

  extract_last<<<(BQ * DQ) / 256, 256, 0, stream>>>(x, (float*)d_out);
}
